// SequenceEncoder_59674275611297
// MI455X (gfx1250) — matrix operand order repaired and checked by reading, not compiled, not run on hardware
//
#include <hip/hip_runtime.h>
#include <hip/hip_bf16.h>

// ---------------------------------------------------------------------------
// Dual forward-scan ConvLSTM encoder for MI455X / gfx1250.
// Conv(3x3, SAME) over concat([x, h]) as shift-decomposed implicit GEMM on
// v_wmma_f32_16x16x32_bf16 (bf16 A/B, fp32 accumulate), fused with LSTM gate
// math and masked state update.
// Padded inputs stored channels-fastest ([B][H+2][W+2][C] bf16) so A-staging
// is pure b128 global->LDS traffic; LDS double-buffered, 1 barrier / K-step.
// ---------------------------------------------------------------------------

typedef __attribute__((ext_vector_type(16))) __bf16    v16bf;
typedef __attribute__((ext_vector_type(8)))  float     v8f;

struct U4x2 { uint4 lo, hi; };

__device__ __forceinline__ unsigned short f2bf(float f) {
    unsigned u = __builtin_bit_cast(unsigned, f);
    return (unsigned short)((u + 0x7fffu + ((u >> 16) & 1u)) >> 16);
}

__device__ __forceinline__ float sigmoidf_(float x) {
    return 1.0f / (1.0f + __expf(-x));
}

// ---------------------------------------------------------------------------
__global__ void zero_kernel(unsigned* __restrict__ p, long long n) {
    long long i = (long long)blockIdx.x * blockDim.x + threadIdx.x;
    if (i < n) p[i] = 0u;
}

// ---------------------------------------------------------------------------
// Pack f32 weights [4C, 2C, 3, 3] -> bf16 [oc4][K], K-order = (kh*3+kw)*2C + ic
// ---------------------------------------------------------------------------
template<int C>
__global__ void pack_weights_kernel(const float* __restrict__ Wsrc,
                                    unsigned short* __restrict__ Wdst) {
    constexpr int K = 18 * C;
    int i = blockIdx.x * 256 + threadIdx.x;
    if (i >= 4 * C * K) return;
    int oc4 = i / K;
    int k   = i - oc4 * K;
    int r9  = k / (2 * C);
    int ic  = k - r9 * (2 * C);
    int kh  = r9 / 3;
    int kw  = r9 - kh * 3;
    Wdst[i] = f2bf(Wsrc[((size_t)(oc4 * (2 * C) + ic) * 3 + kh) * 3 + kw]);
}

// ---------------------------------------------------------------------------
// f32 NCHW -> bf16 zero-padded channels-fastest [B][H+2][W+2][C]
// One thread per padded pixel; contiguous b128 stores.
// ---------------------------------------------------------------------------
template<int C, int H, int W>
__global__ void pad_transpose_kernel(const float* __restrict__ x,
                                     unsigned short* __restrict__ xp) {
    constexpr int PW = W + 2, PH = H + 2;
    int pix = blockIdx.x * 256 + threadIdx.x;       // over 8*PH*PW
    if (pix >= 8 * PH * PW) return;
    int b   = pix / (PH * PW);
    int rem = pix - b * (PH * PW);
    int py  = rem / PW;
    int px  = rem - py * PW;
    unsigned short* dst = xp + (size_t)pix * C;
    const bool interior = (py >= 1 && py <= H && px >= 1 && px <= W);
    const float* srcp = x + (size_t)b * C * H * W + (size_t)(py - 1) * W + (px - 1);
    for (int c8 = 0; c8 < C; c8 += 8) {
        uint4 u = {0u, 0u, 0u, 0u};
        if (interior) {
            unsigned short v[8];
            #pragma unroll
            for (int e = 0; e < 8; ++e)
                v[e] = f2bf(srcp[(size_t)(c8 + e) * (H * W)]);
            u.x = (unsigned)v[0] | ((unsigned)v[1] << 16);
            u.y = (unsigned)v[2] | ((unsigned)v[3] << 16);
            u.z = (unsigned)v[4] | ((unsigned)v[5] << 16);
            u.w = (unsigned)v[6] | ((unsigned)v[7] << 16);
        }
        *(uint4*)(dst + c8) = u;
    }
}

// ---------------------------------------------------------------------------
// Stage one K-step tile (32 channels of one 3x3 shift) into LDS.
// A: thread -> m = tid>>1 (0..63), 16 contiguous channels from (tid&1)*16.
// B: wave = gate, lane = output channel; 64B contiguous copy.
// ---------------------------------------------------------------------------
template<int C, int H, int W>
__device__ __forceinline__ void stage_tile(
    unsigned short* __restrict__ Abuf, unsigned short* __restrict__ Bbuf,
    const unsigned short* __restrict__ xpad, const unsigned short* __restrict__ hpad,
    const unsigned short* __restrict__ Wp, size_t s_row,
    int ks, int b, int y, int a_m, int a_c, int s_g, int s_lane)
{
    constexpr int PW = W + 2, PH = H + 2;
    constexpr int CHUNKS = (2 * C) / 32;

    const int shift = ks / CHUNKS;            // 0..8  (power-of-2 divisor)
    const int icb   = (ks - shift * CHUNKS) * 32;
    const int kh    = shift / 3;
    const int kw    = shift - kh * 3;

    // ---- A: 32B contiguous bf16 load (channels-fastest), 32B LDS store ----
    {
        const int icAll = icb + a_c;
        const unsigned short* src = (icAll < C) ? xpad : hpad;
        const int icp = (icAll < C) ? icAll : icAll - C;
        const int yy = y + (a_m / W) + kh;    // constexpr divisor -> shift
        const int xx = (a_m % W) + kw;
        const uint4* gp = (const uint4*)(src + ((size_t)(b * PH + yy) * PW + xx) * C + icp);
        uint4 q0 = gp[0], q1 = gp[1];
        uint4* dst = (uint4*)&Abuf[a_m * 32 + a_c];
        dst[0] = q0; dst[1] = q1;
    }
    // ---- B: 64B contiguous copy of packed weights ----
    {
        const uint4* src = (const uint4*)(Wp + s_row + (size_t)ks * 32);
        uint4* dst = (uint4*)(Bbuf + s_g * 1024 + s_lane * 32);
        dst[0] = src[0]; dst[1] = src[1]; dst[2] = src[2]; dst[3] = src[3];
    }
}

// ---------------------------------------------------------------------------
// One ConvLSTM timestep, both cells (blockIdx.z). Block = 128 thr = 4 waves
// arranged 2(M) x 2(N); each wave owns two 16-row M-subtiles sharing B frags.
// Block tile: M=64 (row-aligned), N=32 channels per gate, 4 gates.
// ---------------------------------------------------------------------------
template<int C, int H, int W>
__global__ __launch_bounds__(128)
void lstm_step_kernel(const unsigned short* __restrict__ xpad,
                      const int* __restrict__ mask_t,
                      const unsigned short* __restrict__ hpadF, unsigned short* __restrict__ hnpadF,
                      float* __restrict__ cF, float* __restrict__ hF,
                      const unsigned short* __restrict__ WF, const float* __restrict__ bF,
                      const unsigned short* __restrict__ hpadB, unsigned short* __restrict__ hnpadB,
                      float* __restrict__ cB, float* __restrict__ hB,
                      const unsigned short* __restrict__ WB, const float* __restrict__ bB)
{
    constexpr int HW = H * W;
    constexpr int PW = W + 2, PH = H + 2;
    constexpr int K  = 18 * C;
    constexpr int NS = K / 32;

    __shared__ __align__(16) unsigned short Alds[2][64 * 32];
    __shared__ __align__(16) unsigned short Blds[2][4 * 32 * 32];

    const int tid = threadIdx.x;

    const unsigned short* hpad; unsigned short* hnpad; float* cc; float* hfl;
    const unsigned short* Wp; const float* bias;
    if (blockIdx.z == 0) { hpad = hpadF; hnpad = hnpadF; cc = cF; hfl = hF; Wp = WF; bias = bF; }
    else                 { hpad = hpadB; hnpad = hnpadB; cc = cB; hfl = hB; Wp = WB; bias = bB; }

    // Block tile: 64 consecutive m; since 64|m0 and W in {32,64}: x0 == 0.
    const int m0  = blockIdx.x * 64;
    const int b   = m0 / HW;               // constexpr -> shift
    const int rem = m0 - b * HW;
    const int y   = rem / W;               // constexpr -> shift

    const int a_m = tid >> 1;              // 0..63
    const int a_c = (tid & 1) * 16;        // 0 or 16
    const int s_g    = tid >> 5;
    const int s_lane = tid & 31;
    const size_t s_row = (size_t)(s_g * C + blockIdx.y * 32 + s_lane) * (size_t)K;

    const int wave = tid >> 5;
    const int mw = wave >> 1, nw = wave & 1;
    const int lane = tid & 31;
    const int lr = lane & 15;
    const int lh = lane >> 4;

    v8f acc[4][2] = {};

    stage_tile<C, H, W>(Alds[0], Blds[0], xpad, hpad, Wp, s_row, 0,
                        b, y, a_m, a_c, s_g, s_lane);

    for (int ks = 0; ks < NS; ++ks) {
        __syncthreads();
        const int buf = ks & 1;
        if (ks + 1 < NS) {
            stage_tile<C, H, W>(Alds[buf ^ 1], Blds[buf ^ 1], xpad, hpad, Wp, s_row,
                                ks + 1, b, y, a_m, a_c, s_g, s_lane);
        }

        // Two A fragments (ISA 16-bit 16x32 layout): 2 x b128 each
        v16bf a[2];
        #pragma unroll
        for (int s = 0; s < 2; ++s) {
            const unsigned short* arow = &Alds[buf][(mw * 32 + s * 16 + lr) * 32];
            U4x2 t;
            t.lo = *(const uint4*)(arow + lh * 8);
            t.hi = *(const uint4*)(arow + lh * 8 + 16);
            a[s] = __builtin_bit_cast(v16bf, t);
        }

        #pragma unroll
        for (int g = 0; g < 4; ++g) {
            const unsigned short* brow = &Blds[buf][g * 1024 + (nw * 16 + lr) * 32];
            U4x2 t;
            t.lo = *(const uint4*)(brow + lh * 8);
            t.hi = *(const uint4*)(brow + lh * 8 + 16);
            v16bf bfr = __builtin_bit_cast(v16bf, t);
            acc[g][0] = __builtin_amdgcn_wmma_f32_16x16x32_bf16(false, a[0], false, bfr, (short)0, acc[g][0], false, false);
            acc[g][1] = __builtin_amdgcn_wmma_f32_16x16x32_bf16(false, a[1], false, bfr, (short)0, acc[g][1], false, false);
        }
    }

    // ---- epilogue: bias, gates, masked update ----
    const int oc  = blockIdx.y * 32 + nw * 16 + lr;
    const float bi  = bias[oc];
    const float bff = bias[C + oc];
    const float bo  = bias[2 * C + oc];
    const float bg  = bias[3 * C + oc];
    const bool  keep = mask_t[b] > 0;

    #pragma unroll
    for (int s = 0; s < 2; ++s) {
        const int mOff = mw * 32 + s * 16 + lh * 8;       // 8 | W, so one row
        const int yy = y + mOff / W;                      // constexpr -> shift
        const int xx = mOff % W;
        const size_t fbase = ((size_t)(b * C + oc) * H + yy) * W + xx;
        const size_t pbase = ((size_t)(b * PH + yy + 1) * PW + (xx + 1)) * C + oc;
        #pragma unroll
        for (int r = 0; r < 8; ++r) {
            const size_t idx = fbase + r;
            float co  = cc[idx];
            float gi  = sigmoidf_(acc[0][s][r] + bi);
            float gf  = sigmoidf_(acc[1][s][r] + bff);
            float go  = sigmoidf_(acc[2][s][r] + bo);
            float gg  = tanhf(acc[3][s][r] + bg);
            float cn  = gf * co + gi * gg;
            float hnv = go * tanhf(cn);
            float cw  = keep ? cn  : co;
            float hw  = keep ? hnv : hfl[idx];
            cc[idx]  = cw;
            hfl[idx] = hw;
            hnpad[pbase + (size_t)r * C] = f2bf(hw);
        }
    }
}

// ---------------------------------------------------------------------------
__global__ void finalize_kernel(const float* __restrict__ a, const float* __restrict__ b,
                                float* __restrict__ o, int n) {
    int i = blockIdx.x * blockDim.x + threadIdx.x;
    if (i < n) o[i] = 0.5f * (a[i] + b[i]);
}

// ---------------------------------------------------------------------------
extern "C" void kernel_launch(void* const* d_in, const int* in_sizes, int n_in,
                              void* d_out, int out_size, void* d_ws, size_t ws_size,
                              hipStream_t stream) {
    (void)in_sizes; (void)n_in; (void)out_size; (void)ws_size;

    const float* feat0 = (const float*)d_in[0];
    const float* feat1 = (const float*)d_in[1];
    const int*   mask  = (const int*)d_in[2];
    const float* Wf0 = (const float*)d_in[3]; const float* bf0 = (const float*)d_in[4];
    const float* Wb0 = (const float*)d_in[5]; const float* bb0 = (const float*)d_in[6];
    const float* Wf1 = (const float*)d_in[7]; const float* bf1 = (const float*)d_in[8];
    const float* Wb1 = (const float*)d_in[9]; const float* bb1 = (const float*)d_in[10];
    float* out = (float*)d_out;

    const size_t S0 = (size_t)8 * 64 * 64 * 64;     // flat elems, level 0
    const size_t S1 = (size_t)8 * 128 * 32 * 32;    // flat elems, level 1
    const size_t P0 = (size_t)8 * 66 * 66 * 64;     // padded elems, level 0
    const size_t P1 = (size_t)8 * 34 * 34 * 128;    // padded elems, level 1
    const size_t WP0 = (size_t)4 * 64 * (18 * 64);
    const size_t WP1 = (size_t)4 * 128 * (18 * 128);

    auto al = [](size_t s) { return (s + 255) & ~(size_t)255; };
    char* p = (char*)d_ws;
    char* ws0 = p;
    // flat f32 state: c + hflat for 2 cells x 2 levels
    float* cf0 = (float*)p; p += al(S0 * 4);
    float* cb0 = (float*)p; p += al(S0 * 4);
    float* hf0 = (float*)p; p += al(S0 * 4);
    float* hb0 = (float*)p; p += al(S0 * 4);
    float* cf1 = (float*)p; p += al(S1 * 4);
    float* cb1 = (float*)p; p += al(S1 * 4);
    float* hf1 = (float*)p; p += al(S1 * 4);
    float* hb1 = (float*)p; p += al(S1 * 4);
    // padded bf16 h (double buffered per cell) + x, channels-fastest
    unsigned short* hpfA0 = (unsigned short*)p; p += al(P0 * 2);
    unsigned short* hpfB0 = (unsigned short*)p; p += al(P0 * 2);
    unsigned short* hpbA0 = (unsigned short*)p; p += al(P0 * 2);
    unsigned short* hpbB0 = (unsigned short*)p; p += al(P0 * 2);
    unsigned short* xpad0 = (unsigned short*)p; p += al(P0 * 2);
    unsigned short* hpfA1 = (unsigned short*)p; p += al(P1 * 2);
    unsigned short* hpfB1 = (unsigned short*)p; p += al(P1 * 2);
    unsigned short* hpbA1 = (unsigned short*)p; p += al(P1 * 2);
    unsigned short* hpbB1 = (unsigned short*)p; p += al(P1 * 2);
    unsigned short* xpad1 = (unsigned short*)p; p += al(P1 * 2);
    // packed bf16 weights
    unsigned short* Wpf0 = (unsigned short*)p; p += al(WP0 * 2);
    unsigned short* Wpb0 = (unsigned short*)p; p += al(WP0 * 2);
    unsigned short* Wpf1 = (unsigned short*)p; p += al(WP1 * 2);
    unsigned short* Wpb1 = (unsigned short*)p; p += al(WP1 * 2);

    // Zero whole state region (h0 = c0 = 0, padded borders = 0)
    {
        long long nz = (long long)((p - ws0) / 4);
        int grid = (int)((nz + 255) / 256);
        zero_kernel<<<grid, 256, 0, stream>>>((unsigned*)ws0, nz);
    }
    pack_weights_kernel<64 ><<<(int)((WP0 + 255) / 256), 256, 0, stream>>>(Wf0, Wpf0);
    pack_weights_kernel<64 ><<<(int)((WP0 + 255) / 256), 256, 0, stream>>>(Wb0, Wpb0);
    pack_weights_kernel<128><<<(int)((WP1 + 255) / 256), 256, 0, stream>>>(Wf1, Wpf1);
    pack_weights_kernel<128><<<(int)((WP1 + 255) / 256), 256, 0, stream>>>(Wb1, Wpb1);

    unsigned short *pf0 = hpfA0, *nf0 = hpfB0, *pb0 = hpbA0, *nb0 = hpbB0;
    unsigned short *pf1 = hpfA1, *nf1 = hpfB1, *pb1 = hpbA1, *nb1 = hpbB1;

    const int PIX0 = 8 * 66 * 66, PIX1 = 8 * 34 * 34;

    for (int t = 0; t < 16; ++t) {
        const int* mt = mask + t * 8;
        pad_transpose_kernel<64, 64, 64><<<(PIX0 + 255) / 256, 256, 0, stream>>>(
            feat0 + (size_t)t * S0, xpad0);
        pad_transpose_kernel<128, 32, 32><<<(PIX1 + 255) / 256, 256, 0, stream>>>(
            feat1 + (size_t)t * S1, xpad1);

        lstm_step_kernel<64, 64, 64><<<dim3(512, 2, 2), 128, 0, stream>>>(
            xpad0, mt,
            pf0, nf0, cf0, hf0, Wpf0, bf0,
            pb0, nb0, cb0, hb0, Wpb0, bb0);
        lstm_step_kernel<128, 32, 32><<<dim3(128, 4, 2), 128, 0, stream>>>(
            xpad1, mt,
            pf1, nf1, cf1, hf1, Wpf1, bf1,
            pb1, nb1, cb1, hb1, Wpb1, bb1);

        unsigned short* tp;
        tp = pf0; pf0 = nf0; nf0 = tp;
        tp = pb0; pb0 = nb0; nb0 = tp;
        tp = pf1; pf1 = nf1; nf1 = tp;
        tp = pb1; pb1 = nb1; nb1 = tp;
    }

    finalize_kernel<<<(int)((S0 + 255) / 256), 256, 0, stream>>>(hf0, hb0, out, (int)S0);
    finalize_kernel<<<(int)((S1 + 255) / 256), 256, 0, stream>>>(hf1, hb1, out + S0, (int)S1);
}
